// R3Conv_30829275251153
// MI455X (gfx1250) — matrix operand order repaired and checked by reading, not compiled, not run on hardware
//
#include <hip/hip_runtime.h>
#include <hip/hip_bf16.h>

// ---------------------------------------------------------------------------
// Problem constants (from reference)
// ---------------------------------------------------------------------------
#define NBATCH  32
#define NSITES  32768
#define NG      48
#define N_NGB   13
#define DIM     3
#define NSHELLS 16
#define KDIM    (NG * DIM)      // 144
#define KPAD    160             // 5 chunks of K=32 for WMMA; 160 halves/row

typedef __attribute__((ext_vector_type(16))) _Float16 v16h;
typedef __attribute__((ext_vector_type(8)))  _Float16 v8h;   // 16 bytes
typedef __attribute__((ext_vector_type(8)))  float    v8f;

// ---------------------------------------------------------------------------
// Kernel 1: zero the 32x3 output (harness poisons d_out; we atomically
// accumulate into it, so it must start at 0 every call).
// ---------------------------------------------------------------------------
__global__ void k_zero(float* out, int n) {
    int i = blockIdx.x * blockDim.x + threadIdx.x;
    if (i < n) out[i] = 0.0f;
}

// ---------------------------------------------------------------------------
// Kernel 2: W_eff = G_red(3x144) @ wt_perm(144x13), scaled by 1/(Ng*NSites),
// computed with V_WMMA_F32_16X16X32_F16 on a single wave32.
//   G_red[d][j]   = sum_g gdiags[(g*3+d)*144 + j]
//   wt_perm[k][n] = wtVC[k%3][ GnnPerms[k/3][n] ]
//
// LDS staging is f16, pre-padded:
//   sA[16][160] row-major  : rows 0..2 = G_red, rest zero
//   sB[16][160] = wt_perm^T: rows 0..12 = columns of wt_perm, rest zero
// Fragment builds are 2x ds_load_b128 each (ISA layouts are contiguous in K
// per lane): A elem e -> K = h*8+e (e<8) / 16+h*8+(e-8); B elem e -> same K as A.
// ---------------------------------------------------------------------------
__global__ __launch_bounds__(32) void k_weff_wmma(const float* __restrict__ gdiags,
                                                  const int*   __restrict__ perms,
                                                  const float* __restrict__ wtVC,
                                                  float*       __restrict__ weff) {
    __shared__ __align__(16) _Float16 sA[16 * KPAD];   // 5120 B
    __shared__ __align__(16) _Float16 sB[16 * KPAD];   // 5120 B

    const int lane = threadIdx.x;           // 0..31, one wave, EXEC all ones

    // zero-fill both staging buffers (pads included)
    for (int i = lane; i < 16 * KPAD; i += 32) { sA[i] = (_Float16)0.0f; sB[i] = (_Float16)0.0f; }

    // A rows d<3: G_red[d][j] = sum_g gdiags[(g*3+d)*144 + j]
    for (int i = lane; i < DIM * KDIM; i += 32) {
        int d = i / KDIM, j = i % KDIM;
        float s = 0.0f;
        #pragma unroll 4
        for (int g = 0; g < NG; ++g)
            s += gdiags[(g * DIM + d) * KDIM + j];
        sA[d * KPAD + j] = (_Float16)s;
    }

    // B^T rows n<13: sB[n][k] = wt_perm[k][n] = wtVC[k%3][GnnPerms[k/3][n]]
    for (int i = lane; i < KDIM * N_NGB; i += 32) {
        int k = i / N_NGB, n = i % N_NGB;
        int g = k / DIM, d = k % DIM;
        sB[n * KPAD + k] = (_Float16)wtVC[d * N_NGB + perms[g * N_NGB + n]];
    }
    __syncthreads();

    const int m = lane & 15;    // A row / B^T row (= column n) / D col
    const int h = lane >> 4;    // lane half

    v8f c = {};
    #pragma unroll
    for (int kc = 0; kc < 5; ++kc) {
        const int K0 = kc * 32;
        // A: halves e0..7 at K0+h*8, e8..15 at K0+16+h*8 (both 16B aligned)
        v8h alo = *(const v8h*)&sA[m * KPAD + K0 + h * 8];
        v8h ahi = *(const v8h*)&sA[m * KPAD + K0 + 16 + h * 8];
        v16h a = __builtin_shufflevector(alo, ahi,
                 0, 1, 2, 3, 4, 5, 6, 7, 8, 9, 10, 11, 12, 13, 14, 15);
        // B: halves e0..7 at K0+h*8, e8..15 at K0+16+h*8 in B^T row n=m
        v8h blo = *(const v8h*)&sB[m * KPAD + K0 + h * 8];
        v8h bhi = *(const v8h*)&sB[m * KPAD + K0 + 16 + h * 8];
        v16h b = __builtin_shufflevector(blo, bhi,
                 0, 1, 2, 3, 4, 5, 6, 7, 8, 9, 10, 11, 12, 13, 14, 15);
        c = __builtin_amdgcn_wmma_f32_16x16x32_f16(
                /*neg_a=*/false, a, /*neg_b=*/false, b,
                /*c_mod=*/(short)0, c, /*reuse_a=*/false, /*reuse_b=*/false);
    }

    // D 16x16 f32: VGPR r holds row M = h*8 + r, col N = lane&15.
    // Rows 0..2 (the d's) live in lanes 0..15 (h==0), VGPRs 0..2.
    const float scale = 1.0f / ((float)NG * (float)NSITES);
    if (h == 0 && m < N_NGB) {
        #pragma unroll
        for (int r = 0; r < DIM; ++r)
            weff[r * N_NGB + m] = c[r] * scale;
    }
}

// ---------------------------------------------------------------------------
// Kernel 3: gather-reduce (the roofline-critical kernel).
//   T[b,n] = sum_s site_w[s] * In[b, NN[n,s]]
//   out[b,d] += sum_n Weff[d,n] * T[b,n]      (Weff pre-scaled)
// Block = (b, site-chunk). Per thread: 13 accumulators; NN loads coalesced
// along s; In[b,*] is a 128KB slice that stays hot in WGP$/L2 (192MB L2
// holds everything; ~115MB of L2-resident traffic total).
// ---------------------------------------------------------------------------
#define GR_THREADS 256
#define GR_CHUNKS  8

__global__ __launch_bounds__(GR_THREADS) void k_gather_reduce(
        const float* __restrict__ In,     // [32][32768]
        const int*   __restrict__ NN,     // [13][32768]
        const int*   __restrict__ s2sh,   // [32768]
        const float* __restrict__ shw,    // [16]
        const float* __restrict__ weff,   // [3][13] pre-scaled
        float*       __restrict__ out) {  // [32][3]
    const int b   = blockIdx.x;
    const int tid = threadIdx.x;
    const float* __restrict__ Inb = In + (size_t)b * NSITES;

    float acc[N_NGB];
    #pragma unroll
    for (int n = 0; n < N_NGB; ++n) acc[n] = 0.0f;

    const int stride = gridDim.y * GR_THREADS;
    for (int s = blockIdx.y * GR_THREADS + tid; s < NSITES; s += stride) {
        const float ws = shw[s2sh[s]];
        #pragma unroll
        for (int n = 0; n < N_NGB; ++n) {
            const int j = NN[n * NSITES + s];       // coalesced along s
            acc[n] += ws * Inb[j];                  // gather, L2-resident
        }
    }

    // wave32 reduction
    #pragma unroll
    for (int n = 0; n < N_NGB; ++n) {
        #pragma unroll
        for (int off = 16; off >= 1; off >>= 1)
            acc[n] += __shfl_down(acc[n], off, 32);
    }

    __shared__ float part[GR_THREADS / 32][16];
    __shared__ float tot[16];
    const int wave = tid >> 5, lane = tid & 31;
    if (lane == 0) {
        #pragma unroll
        for (int n = 0; n < N_NGB; ++n) part[wave][n] = acc[n];
    }
    __syncthreads();
    if (tid < N_NGB) {
        float t = 0.0f;
        #pragma unroll
        for (int w = 0; w < GR_THREADS / 32; ++w) t += part[w][tid];
        tot[tid] = t;
    }
    __syncthreads();
    if (tid < DIM) {
        float s = 0.0f;
        #pragma unroll
        for (int n = 0; n < N_NGB; ++n) s += weff[tid * N_NGB + n] * tot[n];
        atomicAdd(&out[b * DIM + tid], s);
    }
}

// ---------------------------------------------------------------------------
// Launch: zero -> WMMA W_eff (into d_ws) -> gather-reduce (atomics into d_out)
// Inputs (setup_inputs order): In, NNsites, GnnPerms, SitesToShells, gdiags,
//                              wtVC, ShellWeights
// ---------------------------------------------------------------------------
extern "C" void kernel_launch(void* const* d_in, const int* in_sizes, int n_in,
                              void* d_out, int out_size, void* d_ws, size_t ws_size,
                              hipStream_t stream) {
    const float* In     = (const float*)d_in[0];
    const int*   NN     = (const int*)  d_in[1];
    const int*   perms  = (const int*)  d_in[2];
    const int*   s2sh   = (const int*)  d_in[3];
    const float* gdiags = (const float*)d_in[4];
    const float* wtVC   = (const float*)d_in[5];
    const float* shw    = (const float*)d_in[6];
    float*       out    = (float*)d_out;            // (32, 3)
    float*       weff   = (float*)d_ws;             // 39 floats scratch

    k_zero<<<1, 128, 0, stream>>>(out, out_size);
    k_weff_wmma<<<1, 32, 0, stream>>>(gdiags, perms, wtVC, weff);
    k_gather_reduce<<<dim3(NBATCH, GR_CHUNKS), GR_THREADS, 0, stream>>>(
        In, NN, s2sh, shw, weff, out);
}
